// Attention_25469156066004
// MI455X (gfx1250) — matrix operand order repaired and checked by reading, not compiled, not run on hardware
//
#include <hip/hip_runtime.h>
#include <hip/hip_bf16.h>

// ---------------------------------------------------------------------------
// Fused MHA for MI455X (gfx1250): fp32 -> bf16 WMMA (v_wmma_f32_16x16x32_bf16)
//   B=8, L=1024, D_IN=512, H=8, d=64
//   Pass 0: convert activations + transpose/convert weights to bf16 (mem-bound)
//   Pass 1: projection GEMMs, pure WMMA inner loop (no per-iter conversion)
//   Pass 2: per-wave flash attention, WMMA for QK^T and PV
// ---------------------------------------------------------------------------

typedef __attribute__((ext_vector_type(16))) __bf16 v16bf;
typedef __attribute__((ext_vector_type(8)))  float  v8f;

struct alignas(16) B256 { unsigned w[8]; };   // 32 bytes == v16bf
struct alignas(16) B128 { unsigned w[4]; };   // 16 bytes
struct alignas(8)  B64  { unsigned w[2]; };   // 8 bytes

#define NB_HEAD 8
#define HEAD_D  64
#define SEQ_L   1024
#define D_IN    512
#define D_OUT   512
#define BATCH   8
#define NROWS   (BATCH * SEQ_L)               // 8192

// ---- helpers ---------------------------------------------------------------

__device__ __forceinline__ unsigned f2bf1(float f) {
    unsigned u = __builtin_bit_cast(unsigned, f);
    return (u + 0x7FFFu + ((u >> 16) & 1u)) >> 16;    // RNE fp32 -> bf16
}
__device__ __forceinline__ unsigned packbf(float lo, float hi) {
    return f2bf1(lo) | (f2bf1(hi) << 16);
}
__device__ __forceinline__ v16bf ld_bf16x16(const unsigned short* p) {
    return __builtin_bit_cast(v16bf, *(const B256*)p);
}
__device__ __forceinline__ v16bf ld_bf16x8x2(const unsigned short* p0,
                                             const unsigned short* p1) {
    B256 t;
    B128 a = *(const B128*)p0;
    B128 b = *(const B128*)p1;
    t.w[0] = a.w[0]; t.w[1] = a.w[1]; t.w[2] = a.w[2]; t.w[3] = a.w[3];
    t.w[4] = b.w[0]; t.w[5] = b.w[1]; t.w[6] = b.w[2]; t.w[7] = b.w[3];
    return __builtin_bit_cast(v16bf, t);
}
__device__ __forceinline__ v8f wmma_bf16(v16bf a, v16bf b, v8f c) {
    // D = A(16x32) * B(32x16) + C, f32 accumulate
    return __builtin_amdgcn_wmma_f32_16x16x32_bf16(
        false, a, false, b, (short)0, c, false, false);
}

// ---- kernel 0a: X (fp32) -> Xbf (bf16), same row-major layout --------------
// one thread : 4 elements (float4 in, uint2 out)

__global__ void attn_xconv_kernel(const float* __restrict__ X0,
                                  const float* __restrict__ X1,
                                  const float* __restrict__ X2,
                                  unsigned short* __restrict__ xbf) {
    int id = blockIdx.x * blockDim.x + threadIdx.x;     // 3 * 8192*512/4 threads
    int p  = id >> 20;                                  // / 1048576
    int r  = id & 1048575;
    const float* X = (p == 0) ? X0 : ((p == 1) ? X1 : X2);
    float4 f = *(const float4*)(X + (size_t)r * 4);
    B64 o;
    o.w[0] = packbf(f.x, f.y);
    o.w[1] = packbf(f.z, f.w);
    *(B64*)(xbf + (size_t)p * NROWS * D_IN + (size_t)r * 4) = o;
}

// ---- kernel 0b: W (fp32 [K][N]) -> Wt (bf16 [N][K]) ------------------------

__global__ void attn_wconv_kernel(const float* __restrict__ W0,
                                  const float* __restrict__ W1,
                                  const float* __restrict__ W2,
                                  unsigned short* __restrict__ wt) {
    int id = blockIdx.x * blockDim.x + threadIdx.x;     // 3*512*512 threads
    int p  = id >> 18;
    int r  = id & 262143;
    int n  = r >> 9;
    int k  = r & 511;
    const float* W = (p == 0) ? W0 : ((p == 1) ? W1 : W2);
    wt[p * 262144 + n * 512 + k] = (unsigned short)f2bf1(W[k * 512 + n]);
}

// ---- kernel 1: projections. One wave -> one 16x16 output tile, K=512 -------
// q,k stored bf16 [B,H,L,d]; v stored bf16 transposed [B,H,d,L]

__global__ void attn_proj_kernel(const unsigned short* __restrict__ xbf,
                                 const unsigned short* __restrict__ wt,
                                 unsigned short* __restrict__ qb,
                                 unsigned short* __restrict__ kbuf,
                                 unsigned short* __restrict__ vt) {
    const int p    = blockIdx.y;                         // 0=Q,1=K,2=V
    const int wave = threadIdx.x >> 5;
    const int lane = threadIdx.x & 31;
    const int hi   = lane >> 4;
    const int lnm  = lane & 15;

    const int tile    = blockIdx.x * 8 + wave;           // 0..16383
    const int rowbase = (tile >> 5) * 16;                // 512 row-tiles
    const int nbase   = (tile & 31) * 16;                // 32 col-tiles

    // A row (M = lane%16), bf16 row-major
    const unsigned short* xr = xbf + (size_t)p * NROWS * D_IN
                                   + (size_t)(rowbase + lnm) * D_IN;
    // B "column" row in Wt[n][k] (N = lane%16)
    const unsigned short* wr = wt + p * 262144 + (nbase + lnm) * D_IN;

    v8f c = {};
#pragma unroll 8
    for (int k0 = 0; k0 < D_IN; k0 += 32) {
        // A fragment: elems 0..7 -> k0+hi*8+e, 8..15 -> k0+16+hi*8+e  (two b128)
        v16bf a = ld_bf16x8x2(xr + k0 + hi * 8, xr + k0 + 16 + hi * 8);
        // B fragment: elems 0..7 -> k0+hi*8+e, 8..15 -> k0+16+hi*8+e  (two b128)
        v16bf b = ld_bf16x8x2(wr + k0 + hi * 8, wr + k0 + 16 + hi * 8);
        c = wmma_bf16(a, b, c);
    }

    // C/D layout: lane l, elem v -> (M = v + 8*hi, N = lnm)
#pragma unroll
    for (int v = 0; v < 8; ++v) {
        int r    = rowbase + v + 8 * hi;                 // global seq row
        int col  = nbase + lnm;                          // global output col
        int b    = r >> 10;
        int lpos = r & 1023;
        int h    = col >> 6;
        int d    = col & 63;
        unsigned short val = (unsigned short)f2bf1(c[v]);
        if (p == 0)
            qb[((size_t)(b * NB_HEAD + h) * SEQ_L + lpos) * HEAD_D + d] = val;
        else if (p == 1)
            kbuf[((size_t)(b * NB_HEAD + h) * SEQ_L + lpos) * HEAD_D + d] = val;
        else
            vt[((size_t)(b * NB_HEAD + h) * HEAD_D + d) * SEQ_L + lpos] = val;
    }
}

// ---- kernel 2: per-wave flash attention, 16 query rows, key blocks of 32 ---

__global__ void attn_flash_kernel(const unsigned short* __restrict__ qb,
                                  const unsigned short* __restrict__ kbuf,
                                  const unsigned short* __restrict__ vt,
                                  float* __restrict__ out) {
    __shared__ unsigned short plds[8][512];              // wave-private 16x32 bf16 P tile

    const int wave = threadIdx.x >> 5;
    const int lane = threadIdx.x & 31;
    const int hi   = lane >> 4;
    const int lnm  = lane & 15;

    const int t  = blockIdx.x * 8 + wave;                // 0..4095
    const int bh = t >> 6;                               // (b*H + h)
    const int q0 = (t & 63) * 16;                        // query tile base
    const int b  = bh >> 3;
    const int h  = bh & 7;

    // Q A-fragments (resident all loop): lane row m = lnm, K chunks {0..31},{32..63}
    const unsigned short* qrow = qb + ((size_t)(bh * SEQ_L) + q0 + lnm) * HEAD_D;
    v16bf a0 = ld_bf16x8x2(qrow + 0  + hi * 8, qrow + 16 + hi * 8);
    v16bf a1 = ld_bf16x8x2(qrow + 32 + hi * 8, qrow + 48 + hi * 8);

    v8f o[4] = {};
    float rmax[8], rsum[8];
#pragma unroll
    for (int v = 0; v < 8; ++v) { rmax[v] = -3.0e38f; rsum[v] = 0.0f; }

    const int nblocks = (q0 + 47) >> 5;                  // causal bound, 32-key blocks
    for (int jb = 0; jb < nblocks; ++jb) {
        const int jbase = jb * 32;

        if (jb + 1 < nblocks) {                          // warm next K/V block
            __builtin_prefetch(kbuf + ((size_t)(bh * SEQ_L) + jbase + 32 + lnm) * HEAD_D, 0, 1);
            __builtin_prefetch(vt + ((size_t)(bh * HEAD_D) + lnm) * SEQ_L + jbase + 32, 0, 1);
        }

        // ---- S = Q * K^T  (two 16x16 N-halves, K-dim 64 = 2 WMMAs each) ----
        v8f s0 = {}, s1 = {};
        {
            const unsigned short* kr0 = kbuf + ((size_t)(bh * SEQ_L) + jbase + lnm) * HEAD_D;
            s0 = wmma_bf16(a0, ld_bf16x8x2(kr0 + 0  + hi * 8, kr0 + 16 + hi * 8), s0);
            s0 = wmma_bf16(a1, ld_bf16x8x2(kr0 + 32 + hi * 8, kr0 + 48 + hi * 8), s0);
            const unsigned short* kr1 = kbuf + ((size_t)(bh * SEQ_L) + jbase + 16 + lnm) * HEAD_D;
            s1 = wmma_bf16(a0, ld_bf16x8x2(kr1 + 0  + hi * 8, kr1 + 16 + hi * 8), s1);
            s1 = wmma_bf16(a1, ld_bf16x8x2(kr1 + 32 + hi * 8, kr1 + 48 + hi * 8), s1);
        }

        // ---- scale + causal mask + online softmax --------------------------
        const int jc0 = jbase + lnm;
        const int jc1 = jbase + 16 + lnm;
#pragma unroll
        for (int v = 0; v < 8; ++v) {
            const int row = q0 + v + 8 * hi;
            float x0 = s0[v] * 0.125f; if (jc0 > row) x0 = -1.0e30f;
            float x1 = s1[v] * 0.125f; if (jc1 > row) x1 = -1.0e30f;
            float mx = fmaxf(x0, x1);
            mx = fmaxf(mx, __shfl_xor(mx, 1, 32));
            mx = fmaxf(mx, __shfl_xor(mx, 2, 32));
            mx = fmaxf(mx, __shfl_xor(mx, 4, 32));
            mx = fmaxf(mx, __shfl_xor(mx, 8, 32));
            float nm   = fmaxf(rmax[v], mx);
            float corr = __expf(rmax[v] - nm);
            rmax[v] = nm;
            float e0 = __expf(x0 - nm);
            float e1 = __expf(x1 - nm);
            float sm = e0 + e1;
            sm += __shfl_xor(sm, 1, 32);
            sm += __shfl_xor(sm, 2, 32);
            sm += __shfl_xor(sm, 4, 32);
            sm += __shfl_xor(sm, 8, 32);
            rsum[v] = rsum[v] * corr + sm;
            o[0][v] *= corr; o[1][v] *= corr; o[2][v] *= corr; o[3][v] *= corr;
            // stage P tile (C/D layout -> row-major bf16 in LDS)
            const int m = v + 8 * hi;
            plds[wave][m * 32 + lnm]      = (unsigned short)f2bf1(e0);
            plds[wave][m * 32 + 16 + lnm] = (unsigned short)f2bf1(e1);
        }
        asm volatile("s_wait_dscnt 0" ::: "memory");     // LDS RAW, same wave

        // ---- reload P as A-fragment (16x32) --------------------------------
        const unsigned short* pl = &plds[wave][lnm * 32];
        v16bf pa = ld_bf16x8x2(pl + hi * 8, pl + 16 + hi * 8);

        // ---- O += P * V  (4 d-chunks of 16) --------------------------------
#pragma unroll
        for (int dc = 0; dc < 4; ++dc) {
            const unsigned short* vp =
                vt + ((size_t)(bh * HEAD_D) + dc * 16 + lnm) * SEQ_L + jbase + hi * 8;
            o[dc] = wmma_bf16(pa, ld_bf16x8x2(vp, vp + 16), o[dc]);
        }
    }

    // ---- normalize & store [B,L,H*d] fp32 ---------------------------------
#pragma unroll
    for (int v = 0; v < 8; ++v) {
        float inv  = 1.0f / rsum[v];
        int   lseq = q0 + v + 8 * hi;
        float* orow = out + ((size_t)(b * SEQ_L + lseq)) * D_OUT + h * HEAD_D + lnm;
        orow[0]  = o[0][v] * inv;
        orow[16] = o[1][v] * inv;
        orow[32] = o[2][v] * inv;
        orow[48] = o[3][v] * inv;
    }
}

// ---------------------------------------------------------------------------

extern "C" void kernel_launch(void* const* d_in, const int* in_sizes, int n_in,
                              void* d_out, int out_size, void* d_ws, size_t ws_size,
                              hipStream_t stream) {
    (void)in_sizes; (void)n_in; (void)out_size; (void)ws_size;

    const float* Qs = (const float*)d_in[0];
    const float* Ks = (const float*)d_in[1];
    const float* Vs = (const float*)d_in[2];
    const float* WQ = (const float*)d_in[3];
    const float* WK = (const float*)d_in[4];
    const float* WV = (const float*)d_in[5];
    float* out = (float*)d_out;

    // workspace (bf16 shorts):
    //   Xbf[3][8192][512] | Wt[3][512][512] | q[B,H,L,d] | k[B,H,L,d] | vT[B,H,d,L]
    unsigned short* xbf = (unsigned short*)d_ws;
    unsigned short* wt  = xbf + (size_t)3 * NROWS * D_IN;
    unsigned short* qb  = wt  + (size_t)3 * 512 * 512;
    unsigned short* kb  = qb  + (size_t)BATCH * NB_HEAD * SEQ_L * HEAD_D;
    unsigned short* vt  = kb  + (size_t)BATCH * NB_HEAD * SEQ_L * HEAD_D;

    attn_xconv_kernel<<<12288, 256, 0, stream>>>(Qs, Ks, Vs, xbf);
    attn_wconv_kernel<<<3072, 256, 0, stream>>>(WQ, WK, WV, wt);
    attn_proj_kernel<<<dim3(2048, 3), 256, 0, stream>>>(xbf, wt, qb, kb, vt);
    attn_flash_kernel<<<512, 256, 0, stream>>>(qb, kb, vt, out);
}
